// BlockLTP_44538810860194
// MI455X (gfx1250) — matrix operand order repaired and checked by reading, not compiled, not run on hardware
//
#include <hip/hip_runtime.h>
#include <hip/hip_bf16.h>

// ---------------------------------------------------------------------------
// Transformer block w/ learned token pruning, CDNA5 (gfx1250) WMMA bf16 path.
// B=4, T=2048, C=1024, H=16, HS=64.
// GEMMs: 128x128 block tile, 8 waves, 32x64 wave tile, async LDS staging.
// ---------------------------------------------------------------------------

#define B_ 4
#define T_ 2048
#define C_ 1024
#define H_ 16
#define HS_ 64

typedef unsigned short u16;
typedef __attribute__((ext_vector_type(16))) __bf16 v16bf;
typedef __attribute__((ext_vector_type(8)))  float  v8f;

union Frag { uint4 u4[2]; v16bf v; };

__device__ __forceinline__ u16 f2bf(float f) {
    union { float f; unsigned u; } x; x.f = f;
    unsigned r = x.u + 0x7FFFu + ((x.u >> 16) & 1u);   // round-to-nearest-even
    return (u16)(r >> 16);
}

// CDNA5 async global->LDS copy (ASYNCcnt path, ISA 15.18 op 98)
__device__ __forceinline__ void async_copy_b128(unsigned lds_addr, const void* gaddr) {
    asm volatile("global_load_async_to_lds_b128 %0, %1, off"
                 :: "v"(lds_addr), "v"((unsigned long long)(uintptr_t)gaddr)
                 : "memory");
}
__device__ __forceinline__ void async_wait0() {
    asm volatile("s_wait_asynccnt 0x0" ::: "memory");
}

// ---------------------------------------------------------------------------
// f32 -> bf16 conversion (weights)
// ---------------------------------------------------------------------------
__global__ __launch_bounds__(256)
void cvt_f32_bf16_kernel(const float* __restrict__ src, u16* __restrict__ dst, int n) {
    int i = blockIdx.x * 256 + threadIdx.x;
    if (i < n) dst[i] = f2bf(src[i]);
}

// ---------------------------------------------------------------------------
// LayerNorm over C=1024, output bf16
// ---------------------------------------------------------------------------
__global__ __launch_bounds__(256)
void layernorm_kernel(const float* __restrict__ x, const float* __restrict__ w,
                      const float* __restrict__ bv, u16* __restrict__ out) {
    __shared__ float s1[256], s2[256];
    const int row = blockIdx.x, tid = threadIdx.x;
    const size_t base = (size_t)row * C_;
    float a0 = x[base + tid], a1 = x[base + tid + 256];
    float a2 = x[base + tid + 512], a3 = x[base + tid + 768];
    s1[tid] = a0 + a1 + a2 + a3;
    s2[tid] = a0 * a0 + a1 * a1 + a2 * a2 + a3 * a3;
    __syncthreads();
    for (int off = 128; off > 0; off >>= 1) {
        if (tid < off) { s1[tid] += s1[tid + off]; s2[tid] += s2[tid + off]; }
        __syncthreads();
    }
    const float mu = s1[0] * (1.f / C_);
    const float var = s2[0] * (1.f / C_) - mu * mu;
    const float rs = rsqrtf(var + 1e-5f);
    out[base + tid]       = f2bf((a0 - mu) * rs * w[tid]       + bv[tid]);
    out[base + tid + 256] = f2bf((a1 - mu) * rs * w[tid + 256] + bv[tid + 256]);
    out[base + tid + 512] = f2bf((a2 - mu) * rs * w[tid + 512] + bv[tid + 512]);
    out[base + tid + 768] = f2bf((a3 - mu) * rs * w[tid + 768] + bv[tid + 768]);
}

// ---------------------------------------------------------------------------
// Generic GEMM: D[M,N] = A[M,K](bf16) * W[N,K](bf16)^T + bias, optional GELU,
// output f32 or bf16. Block tile 128x128, 8 waves (4x2), wave tile 32x64.
// LDS tiles staged with CDNA5 async global->LDS copies.
// ---------------------------------------------------------------------------
template <int ACT, int OUTBF>
__global__ __launch_bounds__(256)
void gemm_bt_kernel(const u16* __restrict__ A, const u16* __restrict__ W,
                    const float* __restrict__ bias,
                    float* __restrict__ outF, u16* __restrict__ outB,
                    int M, int N, int K) {
    __shared__ __align__(16) u16 As[128][32];
    __shared__ __align__(16) u16 Bs[128][32];
    const int tid = threadIdx.x;
    const int lane = tid & 31, wave = tid >> 5;
    const int wm = wave & 3, wn = wave >> 2;
    const int mblock = blockIdx.y * 128, nblock = blockIdx.x * 128;
    const int half = lane >> 4, ln = lane & 15;

    v8f acc[2][4];
#pragma unroll
    for (int s = 0; s < 2; ++s)
#pragma unroll
        for (int j = 0; j < 4; ++j)
            acc[s][j] = (v8f){0.f,0.f,0.f,0.f,0.f,0.f,0.f,0.f};

    // staging: each thread copies 32B (2 x b128) of A and of W per K-slab
    const int srow = tid >> 1, skc = (tid & 1) * 16;

    for (int k0 = 0; k0 < K; k0 += 32) {
        const u16* ag = &A[(size_t)(mblock + srow) * K + k0 + skc];
        const u16* wg = &W[(size_t)(nblock + srow) * K + k0 + skc];
        async_copy_b128((unsigned)(uintptr_t)&As[srow][skc], ag);
        async_copy_b128((unsigned)(uintptr_t)&As[srow][skc + 8], ag + 8);
        async_copy_b128((unsigned)(uintptr_t)&Bs[srow][skc], wg);
        async_copy_b128((unsigned)(uintptr_t)&Bs[srow][skc + 8], wg + 8);
        if (k0 + 32 < K) {
            __builtin_prefetch(ag + 32, 0, 0);
            __builtin_prefetch(wg + 32, 0, 0);
        }
        async_wait0();
        __syncthreads();

        Frag af[2];   // A layout: lanes0-15 row=ln k=0..15, lanes16-31 k=16..31
#pragma unroll
        for (int s = 0; s < 2; ++s) {
            const int arr = wm * 32 + s * 16 + ln, akb = half * 16;
            af[s].u4[0] = *reinterpret_cast<const uint4*>(&As[arr][akb]);
            af[s].u4[1] = *reinterpret_cast<const uint4*>(&As[arr][akb + 8]);
        }
#pragma unroll
        for (int j = 0; j < 4; ++j) {
            Frag bf;  // B layout: col=ln, lanes0-15 k=0..15, lanes16-31 k=16..31
            const int brr = wn * 64 + j * 16 + ln;
            const uint4* bp = reinterpret_cast<const uint4*>(&Bs[brr][half * 16]);
            bf.u4[0] = bp[0];
            bf.u4[1] = bp[1];
#pragma unroll
            for (int s = 0; s < 2; ++s)
                acc[s][j] = __builtin_amdgcn_wmma_f32_16x16x32_bf16(
                    false, af[s].v, false, bf.v, (short)0, acc[s][j], false, false);
        }
        __syncthreads();
    }

#pragma unroll
    for (int s = 0; s < 2; ++s) {
        const int mrow0 = mblock + wm * 32 + s * 16 + half * 8;  // D: VGPR r -> M=r (+8 upper)
#pragma unroll
        for (int j = 0; j < 4; ++j) {
            const int col = nblock + wn * 64 + j * 16 + ln;
            const float bb = bias ? bias[col] : 0.f;
#pragma unroll
            for (int r = 0; r < 8; ++r) {
                float v = acc[s][j][r] + bb;
                if (ACT == 1) v = 0.5f * v * (1.0f + erff(v * 0.70710678118654752f));
                const size_t oidx = (size_t)(mrow0 + r) * N + col;
                if (OUTBF) outB[oidx] = f2bf(v);
                else       outF[oidx] = v;
            }
        }
    }
}

// ---------------------------------------------------------------------------
// Repack qkv[M,3C](bf16) -> q,k [B,H,T,HS] and vT [B,H,HS,T]
// ---------------------------------------------------------------------------
__global__ __launch_bounds__(256)
void pack_qkv_kernel(const u16* __restrict__ qkv, u16* __restrict__ q,
                     u16* __restrict__ k, u16* __restrict__ vt) {
    const size_t i = (size_t)blockIdx.x * 256 + threadIdx.x;  // over B*T*C
    const int c = (int)(i & (C_ - 1));
    const size_t row = i >> 10;
    const int b = (int)(row >> 11), t = (int)(row & (T_ - 1));
    const int h = c >> 6, hs = c & 63;
    const size_t bh = (size_t)b * H_ + h;
    q[(bh * T_ + t) * HS_ + hs]  = qkv[row * (3 * C_) + c];
    k[(bh * T_ + t) * HS_ + hs]  = qkv[row * (3 * C_) + C_ + c];
    vt[(bh * HS_ + hs) * T_ + t] = qkv[row * (3 * C_) + 2 * C_ + c];
}

__global__ __launch_bounds__(256)
void zero_f32_kernel(float* __restrict__ p, int n) {
    int i = blockIdx.x * 256 + threadIdx.x;
    if (i < n) p[i] = 0.f;
}

// ---------------------------------------------------------------------------
// Attention: one wave per (b,h,16-query tile). Two-pass online softmax with
// WMMA for QK^T and P.V; importance accumulated via atomics.
// ---------------------------------------------------------------------------
__device__ __forceinline__ void s_tile(const u16* __restrict__ kmat, const Frag* qf,
                                       size_t bhT, int kbs, int half, int ln,
                                       float* sv) {
    Frag kf0, kf1;  // B layout: col(key)=ln, k(d)=half*16+e
    const size_t kb0 = (bhT + (size_t)(kbs + ln)) * HS_ + half * 16;
    kf0.u4[0] = *reinterpret_cast<const uint4*>(&kmat[kb0]);
    kf0.u4[1] = *reinterpret_cast<const uint4*>(&kmat[kb0 + 8]);
    kf1.u4[0] = *reinterpret_cast<const uint4*>(&kmat[kb0 + 32]);
    kf1.u4[1] = *reinterpret_cast<const uint4*>(&kmat[kb0 + 40]);
    v8f s = (v8f){0.f,0.f,0.f,0.f,0.f,0.f,0.f,0.f};
    s = __builtin_amdgcn_wmma_f32_16x16x32_bf16(false, qf[0].v, false, kf0.v, (short)0, s, false, false);
    s = __builtin_amdgcn_wmma_f32_16x16x32_bf16(false, qf[1].v, false, kf1.v, (short)0, s, false, false);
#pragma unroll
    for (int r = 0; r < 8; ++r) sv[r] = s[r] * 0.125f;  // 1/sqrt(64)
}

__global__ __launch_bounds__(32)
void attn_kernel(const u16* __restrict__ qb, const u16* __restrict__ kmat,
                 const u16* __restrict__ vtb, const float* __restrict__ amask,
                 float* __restrict__ imp, u16* __restrict__ attout) {
    __shared__ __align__(16) u16 Plds[16][32];
    const int lane = threadIdx.x;
    const int qt = blockIdx.x & (T_ / 16 - 1);
    const int bh = blockIdx.x >> 7;          // T_/16 = 128
    const int b = bh >> 4, h = bh & 15;
    const int q0 = qt * 16;
    const int half = lane >> 4, ln = lane & 15;
    const size_t bhT = (size_t)bh * T_;

    Frag qf[2];
    {
        const size_t qbase = (bhT + (size_t)(q0 + ln)) * HS_;
        const int kbse = half * 16;
        qf[0].u4[0] = *reinterpret_cast<const uint4*>(&qb[qbase + kbse]);
        qf[0].u4[1] = *reinterpret_cast<const uint4*>(&qb[qbase + kbse + 8]);
        qf[1].u4[0] = *reinterpret_cast<const uint4*>(&qb[qbase + 32 + kbse]);
        qf[1].u4[1] = *reinterpret_cast<const uint4*>(&qb[qbase + 32 + kbse + 8]);
    }

    float m_r[8], l_r[8];
#pragma unroll
    for (int r = 0; r < 8; ++r) { m_r[r] = -1e30f; l_r[r] = 0.f; }

    // ---- pass A: online max / sumexp ----
    for (int kbs = 0; kbs <= q0; kbs += 16) {
        float sv[8];
        s_tile(kmat, qf, bhT, kbs, half, ln, sv);
        const int key = kbs + ln;
        const float am = amask[b * T_ + key];
#pragma unroll
        for (int r = 0; r < 8; ++r) {
            const int qrow = q0 + r + half * 8;
            if (am == 0.f || key > qrow) sv[r] = -1e30f;
        }
#pragma unroll
        for (int r = 0; r < 8; ++r) {
            float mx = sv[r];
            for (int off = 1; off < 16; off <<= 1) mx = fmaxf(mx, __shfl_xor(mx, off));
            const float nm = fmaxf(m_r[r], mx);
            float p = __expf(sv[r] - nm);
            for (int off = 1; off < 16; off <<= 1) p += __shfl_xor(p, off);
            l_r[r] = l_r[r] * __expf(m_r[r] - nm) + p;
            m_r[r] = nm;
        }
    }
    float linv[8];
#pragma unroll
    for (int r = 0; r < 8; ++r) linv[r] = 1.0f / l_r[r];

    // ---- pass B: normalized P, importance, Y = P.V ----
    v8f Y[4];
#pragma unroll
    for (int j = 0; j < 4; ++j) Y[j] = (v8f){0.f,0.f,0.f,0.f,0.f,0.f,0.f,0.f};

    for (int kb = 0; kb <= q0; kb += 32) {
#pragma unroll
        for (int s2 = 0; s2 < 2; ++s2) {
            const int kbs = kb + 16 * s2;
            if (kbs <= q0) {
                float sv[8];
                s_tile(kmat, qf, bhT, kbs, half, ln, sv);
                const int key = kbs + ln;
                const float am = amask[b * T_ + key];
                float part = 0.f;
#pragma unroll
                for (int r = 0; r < 8; ++r) {
                    const int qrow = q0 + r + half * 8;
                    float x = (am == 0.f || key > qrow) ? -1e30f : sv[r];
                    const float p = __expf(x - m_r[r]) * linv[r];
                    part += p;
                    Plds[r + half * 8][s2 * 16 + ln] = f2bf(p);
                }
                part += __shfl_xor(part, 16);
                if (lane < 16) atomicAdd(&imp[b * T_ + kbs + ln], part);
            } else {
#pragma unroll
                for (int r = 0; r < 8; ++r) Plds[r + half * 8][s2 * 16 + ln] = 0;
            }
        }
        __syncthreads();
        Frag pf;  // P as A operand (row=ln, k(keys)=half*16+e)
        pf.u4[0] = *reinterpret_cast<const uint4*>(&Plds[ln][half * 16]);
        pf.u4[1] = *reinterpret_cast<const uint4*>(&Plds[ln][half * 16 + 8]);
#pragma unroll
        for (int j = 0; j < 4; ++j) {
            Frag vf;  // V^T: col d = j*16+ln, k(key)=kb+half*16+e contiguous
            const size_t vb = ((size_t)bh * HS_ + j * 16 + ln) * T_ + kb + half * 16;
            vf.u4[0] = *reinterpret_cast<const uint4*>(&vtb[vb]);
            vf.u4[1] = *reinterpret_cast<const uint4*>(&vtb[vb + 8]);
            Y[j] = __builtin_amdgcn_wmma_f32_16x16x32_bf16(
                false, pf.v, false, vf.v, (short)0, Y[j], false, false);
        }
        __syncthreads();
    }

#pragma unroll
    for (int j = 0; j < 4; ++j)
#pragma unroll
        for (int r = 0; r < 8; ++r) {
            const size_t row = (size_t)b * T_ + q0 + r + half * 8;
            attout[row * C_ + h * HS_ + j * 16 + ln] = f2bf(Y[j][r]);
        }
}

// ---------------------------------------------------------------------------
// residual + pruning mask: x1 = (x + proj) * comb ; write mask to d_out
// ---------------------------------------------------------------------------
__global__ __launch_bounds__(256)
void residual_prune_kernel(const float* __restrict__ x, const float* __restrict__ proj,
                           const float* __restrict__ imp, const float* __restrict__ amask,
                           const float* __restrict__ prot, const float* __restrict__ thr,
                           float* __restrict__ x1, float* __restrict__ mask_out) {
    const int row = blockIdx.x, tid = threadIdx.x;
    const float im = imp[row] * (1.0f / ((float)H_ * (float)T_));
    float pr = (im >= thr[0]) ? 1.f : 0.f;
    if (prot[row] > 0.f) pr = 1.f;
    const float comb = amask[row] * pr;
    const size_t base = (size_t)row * C_;
#pragma unroll
    for (int i = 0; i < 4; ++i) {
        const size_t idx = base + tid + i * 256;
        x1[idx] = (x[idx] + proj[idx]) * comb;
    }
    if (tid == 0) mask_out[row] = comb;
}

__global__ __launch_bounds__(256)
void finalize_kernel(const float* __restrict__ x1, const float* __restrict__ mlp,
                     float* __restrict__ outx, float* __restrict__ loss) {
    const size_t i = (size_t)blockIdx.x * 256 + threadIdx.x;
    outx[i] = x1[i] + mlp[i];
    if (i == 0) loss[0] = 0.f;
}

// ---------------------------------------------------------------------------
// launch
// ---------------------------------------------------------------------------
extern "C" void kernel_launch(void* const* d_in, const int* in_sizes, int n_in,
                              void* d_out, int out_size, void* d_ws, size_t ws_size,
                              hipStream_t stream) {
    (void)in_sizes; (void)n_in; (void)out_size; (void)ws_size;
    const float* x      = (const float*)d_in[0];
    const float* amask  = (const float*)d_in[1];
    const float* prot   = (const float*)d_in[2];
    const float* ln1w   = (const float*)d_in[3];
    const float* ln1b   = (const float*)d_in[4];
    const float* wattn  = (const float*)d_in[5];
    const float* battn  = (const float*)d_in[6];
    const float* wproj  = (const float*)d_in[7];
    const float* bproj  = (const float*)d_in[8];
    const float* ln2w   = (const float*)d_in[9];
    const float* ln2b   = (const float*)d_in[10];
    const float* wfc    = (const float*)d_in[11];
    const float* bfc    = (const float*)d_in[12];
    const float* wfcp   = (const float*)d_in[13];
    const float* bfcp   = (const float*)d_in[14];
    const float* thr    = (const float*)d_in[15];

    char* ws = (char*)d_ws;
    const size_t M = (size_t)B_ * T_;  // 8192
    // workspace layout (bytes)
    u16*   wqkv_bf = (u16*)(ws + 0);                         // 3072*1024
    u16*   wproj_bf= (u16*)(ws + 6291456);                   // 1024*1024
    u16*   wfc_bf  = (u16*)(ws + 8388608);                   // 4096*1024
    u16*   wfcp_bf = (u16*)(ws + 16777216);                  // 1024*4096
    u16*   h_bf    = (u16*)(ws + 25165824);                  // M*1024
    u16*   qkv_bf  = (u16*)(ws + 41943040);                  // M*3072
    u16*   q_bf    = (u16*)(ws + 92274688);                  // B*H*T*64
    u16*   k_bf    = (u16*)(ws + 109051904);
    u16*   vt_bf   = (u16*)(ws + 125829120);                 // B*H*64*T
    float* imp     = (float*)(ws + 142606336);               // B*T
    u16*   att_bf  = (u16*)(ws + 142639104);                 // M*1024
    float* proj    = (float*)(ws + 159416320);               // M*1024 f32
    float* x1      = (float*)(ws + 192970752);               // M*1024 f32
    u16*   h2_bf   = (u16*)(ws + 226525184);                 // M*1024
    u16*   fca_bf  = (u16*)(ws + 243302400);                 // M*4096
    float* mlp     = (float*)(ws + 310411264);               // M*1024 f32

    float* out_x    = (float*)d_out;
    float* out_mask = out_x + M * C_;
    float* out_loss = out_mask + M;

    // 1) weights -> bf16
    cvt_f32_bf16_kernel<<<(3 * C_ * C_) / 256, 256, 0, stream>>>(wattn, wqkv_bf, 3 * C_ * C_);
    cvt_f32_bf16_kernel<<<(C_ * C_) / 256, 256, 0, stream>>>(wproj, wproj_bf, C_ * C_);
    cvt_f32_bf16_kernel<<<(4 * C_ * C_) / 256, 256, 0, stream>>>(wfc, wfc_bf, 4 * C_ * C_);
    cvt_f32_bf16_kernel<<<(4 * C_ * C_) / 256, 256, 0, stream>>>(wfcp, wfcp_bf, 4 * C_ * C_);

    // 2) ln1
    layernorm_kernel<<<(unsigned)M, 256, 0, stream>>>(x, ln1w, ln1b, h_bf);

    // 3) qkv = h @ Wqkv^T + b  (bf16 out)
    gemm_bt_kernel<0, 1><<<dim3(3 * C_ / 128, (unsigned)(M / 128)), 256, 0, stream>>>(
        h_bf, wqkv_bf, battn, nullptr, qkv_bf, (int)M, 3 * C_, C_);

    // 4) repack q/k/vT
    pack_qkv_kernel<<<(unsigned)(M * C_ / 256), 256, 0, stream>>>(qkv_bf, q_bf, k_bf, vt_bf);

    // 5) importance accumulator = 0
    zero_f32_kernel<<<(B_ * T_) / 256, 256, 0, stream>>>(imp, B_ * T_);

    // 6) attention (WMMA flash-style, importance via atomics)
    attn_kernel<<<(unsigned)(B_ * H_ * (T_ / 16)), 32, 0, stream>>>(
        q_bf, k_bf, vt_bf, amask, imp, att_bf);

    // 7) c_proj
    gemm_bt_kernel<0, 0><<<dim3(C_ / 128, (unsigned)(M / 128)), 256, 0, stream>>>(
        att_bf, wproj_bf, bproj, proj, nullptr, (int)M, C_, C_);

    // 8) residual + prune mask
    residual_prune_kernel<<<(unsigned)M, 256, 0, stream>>>(
        x, proj, imp, amask, prot, thr, x1, out_mask);

    // 9) ln2
    layernorm_kernel<<<(unsigned)M, 256, 0, stream>>>(x1, ln2w, ln2b, h2_bf);

    // 10) fc + GELU (bf16 out)
    gemm_bt_kernel<1, 1><<<dim3(4 * C_ / 128, (unsigned)(M / 128)), 256, 0, stream>>>(
        h2_bf, wfc_bf, bfc, nullptr, fca_bf, (int)M, 4 * C_, C_);

    // 11) fcproj
    gemm_bt_kernel<0, 0><<<dim3(C_ / 128, (unsigned)(M / 128)), 256, 0, stream>>>(
        fca_bf, wfcp_bf, bfcp, mlp, nullptr, (int)M, C_, 4 * C_);

    // 12) final residual + loss
    finalize_kernel<<<(unsigned)(M * C_ / 256), 256, 0, stream>>>(x1, mlp, out_x, out_loss);
}
